// Attention_27273042329795
// MI455X (gfx1250) — compile-verified, matrix operand order checked by reading, not run on hardware
//
#include <hip/hip_runtime.h>

typedef __attribute__((ext_vector_type(8)))  float        v8f;
typedef __attribute__((ext_vector_type(8)))  unsigned int v8u;
typedef __attribute__((ext_vector_type(16))) __bf16       v16bf;

__device__ __forceinline__ unsigned short f2bf(float f) {
  union { float f; unsigned int u; } x; x.f = f;
  unsigned int u = x.u + 0x7fffu + ((x.u >> 16) & 1u);
  return (unsigned short)(u >> 16);
}

__device__ __forceinline__ unsigned int pack_bf16(float a, float b) {
  union { float f; unsigned int u; } xa, xb; xa.f = a; xb.f = b;
  unsigned int ua = xa.u + 0x7fffu + ((xa.u >> 16) & 1u);
  unsigned int ub = xb.u + 0x7fffu + ((xb.u >> 16) & 1u);
  return (ua >> 16) | (ub & 0xffff0000u);
}

__device__ __forceinline__ v8f wmma_bf16(v8u a, v8u b, v8f c) {
  return __builtin_amdgcn_wmma_f32_16x16x32_bf16(
      false, __builtin_bit_cast(v16bf, a),
      false, __builtin_bit_cast(v16bf, b),
      (short)0, c, false, false);
}

// ---------------------------------------------------------------------------
// GEMM: C[M,N] = A[M,K] @ W[K,N]  (bf16 WMMA, f32 accumulate)
// MODE 0: A = x (f32), write bf16 Q head-major
// MODE 1: A = x (f32), write bf16 K / V head-major (N = 2048)
// MODE 2: A = attnout (bf16, head-major), write f32 out + bias
// Block tile 128x64, 8 waves in 4x2 grid, each wave 32x32 (2x2 WMMA tiles).
// ---------------------------------------------------------------------------
template <int MODE>
__global__ __launch_bounds__(256) void gemm_kernel(
    const float* __restrict__ Af, const float* __restrict__ W,
    const float* __restrict__ bias, const unsigned int* __restrict__ Abf,
    unsigned short* __restrict__ oQ, unsigned short* __restrict__ oK,
    unsigned short* __restrict__ oV, float* __restrict__ oF,
    int M, int K, int N)
{
  const int tid   = threadIdx.x;
  const int w     = tid >> 5;
  const int lane  = tid & 31;
  const int waveM = w >> 1;       // 0..3
  const int waveN = w & 1;        // 0..1
  const int blockN = blockIdx.x * 64;
  const int blockM = blockIdx.y * 128;

  __shared__ unsigned int lA[128][17];  // [m][kpair]  K=32 per step
  __shared__ unsigned int lB[64][17];   // [n][kpair]  (W transposed)

  v8f acc[2][2] = {};

  for (int kk = 0; kk < K; kk += 32) {
    __syncthreads();
    // ---- A tile: 128 x 32, convert f32 -> packed bf16 pairs ----
    {
      int r = tid >> 1, half = tid & 1;
      int gm = blockM + r;
      if constexpr (MODE != 2) {
        const float2* A2 = (const float2*)Af;
        #pragma unroll
        for (int i = 0; i < 8; ++i) {
          int k = kk + half * 16 + 2 * i;
          float2 f = A2[(size_t)gm * (K >> 1) + (k >> 1)];
          lA[r][half * 8 + i] = pack_bf16(f.x, f.y);
        }
      } else {
        #pragma unroll
        for (int i = 0; i < 8; ++i) {
          int k = kk + half * 16 + 2 * i;
          int b = gm >> 11, nn = gm & 2047, h = k >> 6, d = k & 63;
          lA[r][half * 8 + i] = Abf[(((size_t)(b * 16 + h) * 2048 + nn) * 32) + (d >> 1)];
        }
      }
    }
    // ---- B tile: 32 x 64 of W, stored transposed [n][kpair] ----
    {
      int n = tid & 63, pq = tid >> 6;
      int gn = blockN + n;
      #pragma unroll
      for (int i = 0; i < 4; ++i) {
        int kp = pq * 4 + i;
        int k = kk + 2 * kp;
        lB[n][kp] = pack_bf16(W[(size_t)k * N + gn], W[(size_t)(k + 1) * N + gn]);
      }
    }
    __syncthreads();
    // ---- fragments + WMMA ----
    v8u a0, a1, b0, b1;
    #pragma unroll
    for (int v = 0; v < 8; ++v) {
      int kpi = ((v >= 4) ? 8 : 0) + ((lane >= 16) ? 4 : 0) + (v & 3);
      a0[v] = lA[waveM * 32 + (lane & 15)][kpi];
      a1[v] = lA[waveM * 32 + 16 + (lane & 15)][kpi];
      b0[v] = lB[waveN * 32 + (lane & 15)][kpi];
      b1[v] = lB[waveN * 32 + 16 + (lane & 15)][kpi];
    }
    acc[0][0] = wmma_bf16(a0, b0, acc[0][0]);
    acc[0][1] = wmma_bf16(a0, b1, acc[0][1]);
    acc[1][0] = wmma_bf16(a1, b0, acc[1][0]);
    acc[1][1] = wmma_bf16(a1, b1, acc[1][1]);
  }

  // ---- epilogue ----
  #pragma unroll
  for (int mi = 0; mi < 2; ++mi)
  #pragma unroll
  for (int ni = 0; ni < 2; ++ni)
  #pragma unroll
  for (int v = 0; v < 8; ++v) {
    int row = blockM + waveM * 32 + mi * 16 + v + ((lane >= 16) ? 8 : 0);
    int col = blockN + waveN * 32 + ni * 16 + (lane & 15);
    float val = acc[mi][ni][v];
    if constexpr (MODE == 0) {
      int h = col >> 6, d = col & 63, b = row >> 11, nn = row & 2047;
      oQ[(((size_t)(b * 16 + h) * 2048 + nn) * 64) + d] = f2bf(val);
    } else if constexpr (MODE == 1) {
      int b = row >> 11, nn = row & 2047;
      if (col < 1024) {
        int h = col >> 6, d = col & 63;
        oK[(((size_t)(b * 16 + h) * 2048 + nn) * 64) + d] = f2bf(val);
      } else {
        int c = col - 1024, h = c >> 6, d = c & 63;
        oV[(((size_t)(b * 16 + h) * 2048 + nn) * 64) + d] = f2bf(val);
      }
    } else {
      oF[(size_t)row * N + col] = val + bias[col];
    }
  }
}

// ---------------------------------------------------------------------------
// Attention: per (b*h, 128-row block). FlashAttention-2 online softmax.
// Writes raw dots (scale*QK^T + prev) once, accumulates O = softmax(S) @ V.
// ---------------------------------------------------------------------------
__global__ __launch_bounds__(256) void attn_kernel(
    const unsigned int* __restrict__ Qb,   // bf16 pairs, [bh][n][32]
    const unsigned int* __restrict__ Kbp,  // bf16 pairs, [bh][n][32]
    const unsigned short* __restrict__ Vus,// bf16,       [bh][n][64]
    const float* __restrict__ prev,
    float* __restrict__ dots,
    unsigned short* __restrict__ attnout)  // bf16, [bh][n][64]
{
  const int tid = threadIdx.x, w = tid >> 5, lane = tid & 31;
  const int bh = blockIdx.y;            // 0..31
  const int rowbase = blockIdx.x * 128; // row block

  __shared__ unsigned int   lQ[128][33];
  __shared__ unsigned int   lK[64][33];    // [j][dpair]   (B for Q*K^T)
  __shared__ unsigned int   lVt[64][33];   // [d][jpair]   (B for P*V)
  __shared__ unsigned short lP[8][16][72]; // wave-private P staging

  const size_t qbase = (size_t)bh * 2048 * 32;

  // Q tile -> LDS (already bf16)
  {
    int r = tid >> 1, half = tid & 1;
    #pragma unroll
    for (int i = 0; i < 16; ++i)
      lQ[r][half * 16 + i] = Qb[qbase + (size_t)(rowbase + r) * 32 + half * 16 + i];
  }
  __syncthreads();

  // Q A-fragments, resident for the whole kernel (rows w*16 .. w*16+15)
  v8u qa0, qa1;
  #pragma unroll
  for (int v = 0; v < 8; ++v) {
    int kpi = ((v >= 4) ? 8 : 0) + ((lane >= 16) ? 4 : 0) + (v & 3);
    qa0[v] = lQ[w * 16 + (lane & 15)][kpi];
    qa1[v] = lQ[w * 16 + (lane & 15)][16 + kpi];
  }

  float mrow[8], lrow[8];
  v8f o[4] = {};
  #pragma unroll
  for (int v = 0; v < 8; ++v) { mrow[v] = -3.4e38f; lrow[v] = 0.f; }

  const size_t pbase = (size_t)bh * 2048 * 2048;

  for (int jt = 0; jt < 32; ++jt) {
    const int j = jt * 64;
    __syncthreads();
    // K tile: [j][dpair]
    {
      int r = tid >> 2, q4 = tid & 3;
      #pragma unroll
      for (int i = 0; i < 8; ++i)
        lK[r][q4 * 8 + i] = Kbp[qbase + (size_t)(j + r) * 32 + q4 * 8 + i];
    }
    // V tile transposed: [d][jpair]
    {
      int d = tid & 63, pj = tid >> 6;
      #pragma unroll
      for (int i = 0; i < 8; ++i) {
        int p = pj * 8 + i;
        unsigned int u0 = Vus[((size_t)bh * 2048 + j + 2 * p) * 64 + d];
        unsigned int u1 = Vus[((size_t)bh * 2048 + j + 2 * p + 1) * 64 + d];
        lVt[d][p] = u0 | (u1 << 16);
      }
    }
    __syncthreads();

    // S = Q @ K^T  (wave: 16 rows x 64 cols = 4 WMMA tiles)
    v8f s[4];
    #pragma unroll
    for (int nt = 0; nt < 4; ++nt) {
      v8u kb0, kb1;
      int krow = nt * 16 + (lane & 15);
      #pragma unroll
      for (int v = 0; v < 8; ++v) {
        int kpi = ((v >= 4) ? 8 : 0) + ((lane >= 16) ? 4 : 0) + (v & 3);
        kb0[v] = lK[krow][kpi];
        kb1[v] = lK[krow][16 + kpi];
      }
      v8f a = {};
      a = wmma_bf16(qa0, kb0, a);
      a = wmma_bf16(qa1, kb1, a);
      s[nt] = a;
    }

    // scale + prev, write dots (raw pre-softmax scores are an output)
    const int rlo = rowbase + w * 16 + ((lane >= 16) ? 8 : 0);
    const int c0 = j + (lane & 15);
    #pragma unroll
    for (int nt = 0; nt < 4; ++nt)
    #pragma unroll
    for (int v = 0; v < 8; ++v) {
      size_t idx = pbase + (size_t)(rlo + v) * 2048 + c0 + nt * 16;
      float val = s[nt][v] * 0.125f + prev[idx];
      dots[idx] = val;
      s[nt][v] = val;
    }

    // online softmax (row = v + 8*(lane>=16); reduce over 16-lane groups)
    #pragma unroll
    for (int v = 0; v < 8; ++v) {
      float tm = fmaxf(fmaxf(s[0][v], s[1][v]), fmaxf(s[2][v], s[3][v]));
      tm = fmaxf(tm, __shfl_xor(tm, 1));
      tm = fmaxf(tm, __shfl_xor(tm, 2));
      tm = fmaxf(tm, __shfl_xor(tm, 4));
      tm = fmaxf(tm, __shfl_xor(tm, 8));
      float nm = fmaxf(mrow[v], tm);
      float alpha = __expf(mrow[v] - nm);
      mrow[v] = nm;
      float rs = 0.f;
      #pragma unroll
      for (int nt = 0; nt < 4; ++nt) {
        float p = __expf(s[nt][v] - nm);
        s[nt][v] = p;
        rs += p;
      }
      rs += __shfl_xor(rs, 1);
      rs += __shfl_xor(rs, 2);
      rs += __shfl_xor(rs, 4);
      rs += __shfl_xor(rs, 8);
      lrow[v] = lrow[v] * alpha + rs;
      #pragma unroll
      for (int dt = 0; dt < 4; ++dt) o[dt][v] *= alpha;
    }

    // stage P through wave-private LDS: C-layout -> A-layout
    #pragma unroll
    for (int nt = 0; nt < 4; ++nt)
    #pragma unroll
    for (int v = 0; v < 8; ++v) {
      int rr = v + ((lane >= 16) ? 8 : 0);
      lP[w][rr][nt * 16 + (lane & 15)] = f2bf(s[nt][v]);
    }
    v8u pa0, pa1;
    #pragma unroll
    for (int v = 0; v < 8; ++v) {
      int kpi = ((v >= 4) ? 8 : 0) + ((lane >= 16) ? 4 : 0) + (v & 3);
      const unsigned int* pr = (const unsigned int*)&lP[w][lane & 15][0];
      pa0[v] = pr[kpi];
      pa1[v] = pr[16 + kpi];
    }

    // O += P @ V
    #pragma unroll
    for (int dt = 0; dt < 4; ++dt) {
      v8u vb0, vb1;
      int drow = dt * 16 + (lane & 15);
      #pragma unroll
      for (int v = 0; v < 8; ++v) {
        int kpi = ((v >= 4) ? 8 : 0) + ((lane >= 16) ? 4 : 0) + (v & 3);
        vb0[v] = lVt[drow][kpi];
        vb1[v] = lVt[drow][16 + kpi];
      }
      o[dt] = wmma_bf16(pa0, vb0, o[dt]);
      o[dt] = wmma_bf16(pa1, vb1, o[dt]);
    }
  }

  // normalize and store bf16 for the output projection
  #pragma unroll
  for (int dt = 0; dt < 4; ++dt)
  #pragma unroll
  for (int v = 0; v < 8; ++v) {
    int row = rowbase + w * 16 + v + ((lane >= 16) ? 8 : 0);
    float val = o[dt][v] / lrow[v];
    attnout[((size_t)bh * 2048 + row) * 64 + dt * 16 + (lane & 15)] = f2bf(val);
  }
}

// ---------------------------------------------------------------------------
extern "C" void kernel_launch(void* const* d_in, const int* in_sizes, int n_in,
                              void* d_out, int out_size, void* d_ws, size_t ws_size,
                              hipStream_t stream) {
  const float* x    = (const float*)d_in[0];
  const float* prev = (const float*)d_in[1];
  const float* Wq   = (const float*)d_in[2];
  const float* Wkv  = (const float*)d_in[3];
  const float* Wo   = (const float*)d_in[4];
  const float* bo   = (const float*)d_in[5];

  float* out  = (float*)d_out;                       // (2,2048,1024) f32
  float* dots = out + (size_t)2 * 2048 * 1024;       // (2,16,2048,2048) f32

  const size_t HEADEL = (size_t)2 * 16 * 2048 * 64;  // 4,194,304 elements
  unsigned short* Qbf  = (unsigned short*)d_ws;
  unsigned short* Kbf  = Qbf + HEADEL;
  unsigned short* Vbf  = Kbf + HEADEL;
  unsigned short* AObf = Vbf + HEADEL;               // 32 MB total bf16 scratch

  // q = x @ Wq
  gemm_kernel<0><<<dim3(16, 32), 256, 0, stream>>>(
      x, Wq, nullptr, nullptr, Qbf, nullptr, nullptr, nullptr, 4096, 1024, 1024);
  // kv = x @ Wkv  -> K, V
  gemm_kernel<1><<<dim3(32, 32), 256, 0, stream>>>(
      x, Wkv, nullptr, nullptr, nullptr, Kbf, Vbf, nullptr, 4096, 1024, 2048);
  // dots + softmax + attn@v
  attn_kernel<<<dim3(16, 32), 256, 0, stream>>>(
      (const unsigned int*)Qbf, (const unsigned int*)Kbf,
      (const unsigned short*)Vbf, prev, dots, AObf);
  // out = attnout @ Wo + bo
  gemm_kernel<2><<<dim3(16, 32), 256, 0, stream>>>(
      nullptr, Wo, bo, (const unsigned int*)AObf, nullptr, nullptr, nullptr,
      out, 4096, 1024, 1024);
}
